// CausalSelfAttention_82136954568961
// MI455X (gfx1250) — compile-verified, matrix operand order checked by reading, not run on hardware
//
#include <hip/hip_runtime.h>

typedef __attribute__((ext_vector_type(16))) __bf16          v16bf;
typedef __attribute__((ext_vector_type(8)))  float           v8f;
typedef __attribute__((ext_vector_type(8)))  unsigned int    v8u;
typedef __attribute__((ext_vector_type(16))) unsigned short  v16us;

// ---- gfx1250 async direct-to-LDS copy (ASYNCcnt-tracked), with fallback -----
#if defined(__has_builtin)
#if __has_builtin(__builtin_amdgcn_global_load_async_to_lds_b128)
#define HAS_ASYNC_LDS 1
#endif
#endif
#ifndef HAS_ASYNC_LDS
#define HAS_ASYNC_LDS 0
#endif

typedef int v4i_vs __attribute__((vector_size(16)));             // GCC-style int4
typedef __attribute__((address_space(1))) v4i_vs* glob_v4i_ptr;  // __device__ int4*
typedef __attribute__((address_space(3))) v4i_vs* lds_v4i_ptr;   // __shared__ int4*

__device__ __forceinline__ void async_copy16(unsigned short* lds_dst,
                                             const unsigned short* gsrc) {
#if HAS_ASYNC_LDS
  __builtin_amdgcn_global_load_async_to_lds_b128(
      (glob_v4i_ptr)gsrc, (lds_v4i_ptr)lds_dst, 0, 0);
#else
  *(uint4*)lds_dst = *(const uint4*)gsrc;   // VGPR round trip fallback
#endif
}

__device__ __forceinline__ void wait_async_all() {
#if HAS_ASYNC_LDS
#if __has_builtin(__builtin_amdgcn_s_wait_asynccnt)
  __builtin_amdgcn_s_wait_asynccnt(0);
#else
  asm volatile("s_wait_asynccnt 0x0" ::: "memory");
#endif
#endif
}

// fp32 -> bf16 (round-to-nearest-even)
__device__ __forceinline__ unsigned short f2bf(float f) {
  unsigned u = __builtin_bit_cast(unsigned, f);
  u += 0x7FFFu + ((u >> 16) & 1u);
  return (unsigned short)(u >> 16);
}

__device__ __forceinline__ v8f wmma_bf16(v16bf a, v16bf b, v8f c) {
  return __builtin_amdgcn_wmma_f32_16x16x32_bf16(false, a, false, b, (short)0, c,
                                                 false, false);
}

// A-fragment loader, 16x32 bf16 tile with contraction contiguous ([row][k]).
// Also the B loader when B is stored N-major ([n][k]) — layouts mirror.
__device__ __forceinline__ v16bf load_frag_kcontig(const unsigned short* p0, int ld) {
  const int lane = threadIdx.x & 31;
  const unsigned short* p = p0 + (size_t)(lane & 15) * ld + ((lane >> 4) << 3);
  v8u a;
#pragma unroll
  for (int v = 0; v < 4; ++v) a[v] = *(const unsigned int*)(p + 2 * v);
#pragma unroll
  for (int v = 0; v < 4; ++v) a[4 + v] = *(const unsigned int*)(p + 16 + 2 * v);
  return __builtin_bit_cast(v16bf, a);
}

// B-fragment loader for a 32x16 tile stored K-major ([k][n], stride ld).
__device__ __forceinline__ v16bf load_frag_b_kmaj(const unsigned short* p0, int ld) {
  const int lane = threadIdx.x & 31;
  const int half8 = (lane >> 4) << 3;
  const unsigned short* p = p0 + (lane & 15);
  v16us b;
#pragma unroll
  for (int v = 0; v < 8; ++v) {
    int k = ((v < 4) ? (v << 1) : (16 + ((v - 4) << 1))) + half8;
    b[2 * v]     = p[(size_t)k * ld];
    b[2 * v + 1] = p[(size_t)(k + 1) * ld];
  }
  return __builtin_bit_cast(v16bf, b);
}

// ---------------- fp32 -> bf16 pack --------------------------------------
__global__ __launch_bounds__(256) void pack_bf16_kernel(
    const float* __restrict__ src, unsigned short* __restrict__ dst,
    int n, int cols, int ldDst, int colOff) {
  int i = blockIdx.x * 256 + threadIdx.x;
  if (i >= n) return;
  int r = i / cols, c = i - r * cols;
  dst[(size_t)r * ldDst + colOff + c] = f2bf(src[i]);
}

// ---------------- bf16 WMMA GEMM: C[M,N] = A[M,K] * B[K,N] ------------------
// 256 threads = 8 waves (4x2), block tile 256x128, wave tile 64x64 (4x4 WMMA
// tiles -> 16 WMMAs per 8 fragment loads), K step 32, double-buffered LDS with
// async direct-to-LDS staging of tile k+1 overlapped with compute on tile k.
template <bool OUT_BF16>
__global__ __launch_bounds__(256) void gemm_bf16_kernel(
    const unsigned short* __restrict__ A, const unsigned short* __restrict__ B,
    void* __restrict__ Cout, int M, int N, int K) {
  __shared__ __align__(16) unsigned short As[2][256 * 40];  // [m][k], stride 40
  __shared__ __align__(16) unsigned short Bs[2][32 * 136];  // [k][n], stride 136
  const int tid = threadIdx.x;
  const int w = tid >> 5;
  const int wm = w >> 1;   // 0..3 (64-row tiles)
  const int wn = w & 1;    // 0..1 (64-col tiles)
  const int m0 = blockIdx.y * 256;
  const int n0 = blockIdx.x * 128;

  v8f acc[4][4] = {};

  // staging in 16B chunks: A 4 chunks/row (64 rows/pass x4), B 16 chunks/row
  const int arow = tid >> 2, ac8 = (tid & 3) << 3;
  const int brow = tid >> 4, bc8 = (tid & 15) << 3;

  auto stage = [&](int buf, int k0) {
#pragma unroll
    for (int p = 0; p < 4; ++p) {
      int ra = arow + p * 64;
      async_copy16(&As[buf][ra * 40 + ac8],
                   A + (size_t)(m0 + ra) * K + k0 + ac8);
    }
#pragma unroll
    for (int p = 0; p < 2; ++p) {
      int rb = brow + p * 16;
      async_copy16(&Bs[buf][rb * 136 + bc8],
                   B + (size_t)(k0 + rb) * N + n0 + bc8);
    }
  };

  stage(0, 0);
  wait_async_all();
  __syncthreads();

  for (int k0 = 0; k0 < K; k0 += 32) {
    const int cur = (k0 >> 5) & 1;
    if (k0 + 32 < K) stage(cur ^ 1, k0 + 32);  // overlap next tile with compute
    v16bf af[4];
#pragma unroll
    for (int i = 0; i < 4; ++i)
      af[i] = load_frag_kcontig(&As[cur][(wm * 64 + i * 16) * 40], 40);
#pragma unroll
    for (int j = 0; j < 4; ++j) {
      v16bf bfrag = load_frag_b_kmaj(&Bs[cur][wn * 64 + j * 16], 136);
#pragma unroll
      for (int i = 0; i < 4; ++i) acc[i][j] = wmma_bf16(af[i], bfrag, acc[i][j]);
    }
    wait_async_all();
    __syncthreads();
  }

  const int lane = tid & 31;
  const int col = lane & 15, rb8 = (lane >> 4) << 3;
#pragma unroll
  for (int i = 0; i < 4; ++i) {
#pragma unroll
    for (int j = 0; j < 4; ++j) {
      int m = m0 + wm * 64 + i * 16 + rb8;
      int n = n0 + wn * 64 + j * 16 + col;
#pragma unroll
      for (int r = 0; r < 8; ++r) {
        float v = acc[i][j][r];
        if (OUT_BF16)
          ((unsigned short*)Cout)[(size_t)(m + r) * N + n] = f2bf(v);
        else
          ((float*)Cout)[(size_t)(m + r) * N + n] = v;
      }
    }
  }
}

// ---------------- fused causal flash attention ------------------------------
// qkv: [B*T, 3072] bf16 (Q | K | V, head h = 64 consecutive cols).
// Grid = B*H*(T/64) blocks, 128 threads (4 waves); wave w owns query rows
// [q0, q0+15]. Key/value tiles of 32 double-buffered in LDS (async staged).
__global__ __launch_bounds__(128) void attn_kernel(
    const unsigned short* __restrict__ qkv, unsigned short* __restrict__ att) {
  constexpr int T = 1024, CQ = 3072, LDK = 72, LDP = 40;
  __shared__ __align__(16) unsigned short Ks[2][32 * LDK];
  __shared__ __align__(16) unsigned short Vs[2][32 * LDK];
  __shared__ __align__(16) unsigned short Ps[4][16 * LDP];

  const int tid = threadIdx.x;
  const int w = tid >> 5, lane = tid & 31;
  const int col = lane & 15, half = lane >> 4;
  const int qblk = blockIdx.x & 15;
  const int h = (blockIdx.x >> 4) & 15;
  const int b = blockIdx.x >> 8;
  const int q0 = qblk * 64 + w * 16;

  const unsigned short* qbase = qkv + ((size_t)b * T + q0) * CQ + h * 64;
  const unsigned short* kbase = qkv + (size_t)b * T * CQ + 1024 + h * 64;
  const unsigned short* vbase = kbase + 1024;

  v16bf qf0 = load_frag_kcontig(qbase, CQ);        // Q kept in registers
  v16bf qf1 = load_frag_kcontig(qbase + 32, CQ);

  v8f o[4] = {};
  float m_i[8], l_i[8];
#pragma unroll
  for (int r = 0; r < 8; ++r) { m_i[r] = -1e30f; l_i[r] = 0.f; }

  // staging: 8 chunks/row of 16B, 16 rows/pass x2
  const int srow = tid >> 3, sc8 = (tid & 7) << 3;
  auto stage = [&](int buf, int key0) {
#pragma unroll
    for (int p = 0; p < 2; ++p) {
      int r = srow + p * 16;
      async_copy16(&Ks[buf][r * LDK + sc8],
                   kbase + (size_t)(key0 + r) * CQ + sc8);
      async_copy16(&Vs[buf][r * LDK + sc8],
                   vbase + (size_t)(key0 + r) * CQ + sc8);
    }
  };

  const int ktmax = (qblk * 64 + 63) >> 5;  // causal upper bound
  stage(0, 0);
  wait_async_all();
  __syncthreads();

  for (int kt = 0; kt <= ktmax; ++kt) {
    const int key0 = kt << 5;
    const int cur = kt & 1;
    if (kt < ktmax) stage(cur ^ 1, (kt + 1) << 5);  // overlap next K/V tile
    const bool active = (key0 <= q0 + 15);
    if (active) {
      float sv[2][8];
#pragma unroll
      for (int j = 0; j < 2; ++j) {
        // S(16x16) = Q(16x64) * K^T; Ks is [key][d] -> K^T is N-major
        v8f s = {};
        s = wmma_bf16(qf0, load_frag_kcontig(&Ks[cur][(j * 16) * LDK], LDK), s);
        s = wmma_bf16(qf1, load_frag_kcontig(&Ks[cur][(j * 16) * LDK + 32], LDK), s);
#pragma unroll
        for (int r = 0; r < 8; ++r) {
          int keyg = key0 + j * 16 + col;
          int qg = q0 + r + 8 * half;
          sv[j][r] = (keyg <= qg) ? s[r] * 0.125f : -1e30f;  // 1/sqrt(64) + mask
        }
      }
      // online softmax; row stats via 16-lane butterfly (cols across lanes)
#pragma unroll
      for (int r = 0; r < 8; ++r) {
        float rm = fmaxf(sv[0][r], sv[1][r]);
#pragma unroll
        for (int off = 1; off < 16; off <<= 1)
          rm = fmaxf(rm, __shfl_xor(rm, off, 32));
        float mn = fmaxf(m_i[r], rm);
        float fc = __expf(m_i[r] - mn);
        float pa = __expf(sv[0][r] - mn);
        float pb = __expf(sv[1][r] - mn);
        float rs = pa + pb;
#pragma unroll
        for (int off = 1; off < 16; off <<= 1) rs += __shfl_xor(rs, off, 32);
        l_i[r] = l_i[r] * fc + rs;
        m_i[r] = mn;
        Ps[w][(r + 8 * half) * LDP + col]      = f2bf(pa);  // C-layout -> LDS
        Ps[w][(r + 8 * half) * LDP + 16 + col] = f2bf(pb);
#pragma unroll
        for (int n = 0; n < 4; ++n) o[n][r] *= fc;
      }
    }
    __syncthreads();
    if (active) {
      v16bf pf = load_frag_kcontig(&Ps[w][0], LDP);  // P as 16x32 A matrix
#pragma unroll
      for (int n = 0; n < 4; ++n)                    // O(16x64) += P * V(32x64)
        o[n] = wmma_bf16(pf, load_frag_b_kmaj(&Vs[cur][n * 16], LDK), o[n]);
    }
    wait_async_all();
    __syncthreads();
  }
#pragma unroll
  for (int r = 0; r < 8; ++r) {
    float invl = 1.0f / l_i[r];
#pragma unroll
    for (int n = 0; n < 4; ++n) {
      size_t idx = ((size_t)b * T + q0 + r + 8 * half) * 1024 + h * 64 + n * 16 + col;
      att[idx] = f2bf(o[n][r] * invl);
    }
  }
}

extern "C" void kernel_launch(void* const* d_in, const int* in_sizes, int n_in,
                              void* d_out, int out_size, void* d_ws, size_t ws_size,
                              hipStream_t stream) {
  const float* x  = (const float*)d_in[0];
  const float* Wq = (const float*)d_in[1];
  const float* Wk = (const float*)d_in[2];
  const float* Wv = (const float*)d_in[3];
  const float* Wp = (const float*)d_in[4];
  float* out = (float*)d_out;

  // workspace layout (bf16 = unsigned short), total 48 MB
  unsigned short* xb    = (unsigned short*)d_ws;
  unsigned short* wqkv  = xb    + (size_t)4096 * 1024;
  unsigned short* wproj = wqkv  + (size_t)1024 * 3072;
  unsigned short* qkv   = wproj + (size_t)1024 * 1024;
  unsigned short* att   = qkv   + (size_t)4096 * 3072;

  const int nX = 4096 * 1024, nW = 1024 * 1024;
  pack_bf16_kernel<<<(nX + 255) / 256, 256, 0, stream>>>(x,  xb,    nX, 1024, 1024, 0);
  pack_bf16_kernel<<<(nW + 255) / 256, 256, 0, stream>>>(Wq, wqkv,  nW, 1024, 3072, 0);
  pack_bf16_kernel<<<(nW + 255) / 256, 256, 0, stream>>>(Wk, wqkv,  nW, 1024, 3072, 1024);
  pack_bf16_kernel<<<(nW + 255) / 256, 256, 0, stream>>>(Wv, wqkv,  nW, 1024, 3072, 2048);
  pack_bf16_kernel<<<(nW + 255) / 256, 256, 0, stream>>>(Wp, wproj, nW, 1024, 1024, 0);

  // fused QKV projection: [4096,1024] x [1024,3072] -> bf16 [4096,3072]
  gemm_bf16_kernel<true><<<dim3(3072 / 128, 4096 / 256), 256, 0, stream>>>(
      xb, wqkv, qkv, 4096, 3072, 1024);

  // fused causal attention: B*H*(T/64) = 1024 blocks
  attn_kernel<<<1024, 128, 0, stream>>>(qkv, att);

  // output projection: [4096,1024] x [1024,1024] -> fp32 d_out
  gemm_bf16_kernel<false><<<dim3(1024 / 128, 4096 / 256), 256, 0, stream>>>(
      att, wproj, out, 4096, 1024, 1024);
}
